// Top2Router_13013750907087
// MI455X (gfx1250) — compile-verified, matrix operand order checked by reading, not run on hardware
//
#include <hip/hip_runtime.h>
#include <math.h>

#define DIM    2048
#define NEXP   64
#define NTOK   16384          // 4 * 4096 tokens
#define KSTEPS 64             // 2048 / 32
#define WPB    8              // waves per block (256 threads, wave32)
#define MTILE  16             // token rows per wave
#define PFDIST 8              // prefetch distance in K-blocks (1 KB per row)

typedef __attribute__((ext_vector_type(16))) __bf16 v16bf;
typedef __attribute__((ext_vector_type(8)))  float  v8f;

// ---------------------------------------------------------------------------
// Kernel 1: convert W [2048 x 64] f32 -> bf16, packed in WMMA B-fragment order.
// Buffer = 256 fragments (kb 0..63  x  ntile 0..3), each fragment 1024 B:
//   [lane 0..31][8 dwords], lane holds column n = ntile*16 + lane%16,
//   K values: klo = (lane>=16 ? 8 : 0); dword j<4 -> K = 32*kb+klo+2j,+1
//                                       dword j>=4 -> K = 32*kb+klo+16+2(j-4),+1
// ---------------------------------------------------------------------------
__global__ __launch_bounds__(256) void pack_w_kernel(const float* __restrict__ W,
                                                     unsigned* __restrict__ wpk) {
    const int d    = blockIdx.x * 256 + threadIdx.x;   // 0..65535 output dwords
    const int frag = d >> 8;
    const int r    = d & 255;
    const int lane = r >> 3;
    const int j    = r & 7;
    const int kb   = frag >> 2;
    const int t    = frag & 3;
    const int n    = t * 16 + (lane & 15);
    const int klo  = (lane & 16) ? 8 : 0;
    const int k    = kb * 32 + klo + ((j < 4) ? (2 * j) : (16 + 2 * (j - 4)));

    const float f0 = W[(size_t)k * NEXP + n];
    const float f1 = W[(size_t)(k + 1) * NEXP + n];
    const __bf16 b0 = (__bf16)f0;
    const __bf16 b1 = (__bf16)f1;
    const unsigned u0 = (unsigned)__builtin_bit_cast(unsigned short, b0);
    const unsigned u1 = (unsigned)__builtin_bit_cast(unsigned short, b1);
    wpk[d] = (u1 << 16) | u0;
}

// ---------------------------------------------------------------------------
// Helpers for the GEMM inner loop.
// ---------------------------------------------------------------------------
__device__ __forceinline__ void load_ablock(const float* __restrict__ p, float4* b) {
    b[0] = *(const float4*)(p + 0);
    b[1] = *(const float4*)(p + 4);
    b[2] = *(const float4*)(p + 16);
    b[3] = *(const float4*)(p + 20);
}

__device__ __forceinline__ v16bf cvt_afrag(const float4* b) {
    v16bf af;
    af[0]  = (__bf16)b[0].x; af[1]  = (__bf16)b[0].y;
    af[2]  = (__bf16)b[0].z; af[3]  = (__bf16)b[0].w;
    af[4]  = (__bf16)b[1].x; af[5]  = (__bf16)b[1].y;
    af[6]  = (__bf16)b[1].z; af[7]  = (__bf16)b[1].w;
    af[8]  = (__bf16)b[2].x; af[9]  = (__bf16)b[2].y;
    af[10] = (__bf16)b[2].z; af[11] = (__bf16)b[2].w;
    af[12] = (__bf16)b[3].x; af[13] = (__bf16)b[3].y;
    af[14] = (__bf16)b[3].z; af[15] = (__bf16)b[3].w;
    return af;
}

__device__ __forceinline__ void wmma_kblock(const unsigned char* __restrict__ smem,
                                            int kb, int lane, const float4* stage,
                                            v8f& acc0, v8f& acc1, v8f& acc2, v8f& acc3) {
    const v16bf af = cvt_afrag(stage);
    const unsigned char* fb = smem + (size_t)(kb * 4) * 1024 + (size_t)lane * 32;
    const v16bf b0 = *(const v16bf*)(fb);
    const v16bf b1 = *(const v16bf*)(fb + 1024);
    const v16bf b2 = *(const v16bf*)(fb + 2048);
    const v16bf b3 = *(const v16bf*)(fb + 3072);
    // Keep all 8 ds_load_b128 issued before the first WMMA so the dscnt waits
    // are staggered instead of a full LDS round-trip per WMMA.
    __builtin_amdgcn_sched_barrier(0);
    acc0 = __builtin_amdgcn_wmma_f32_16x16x32_bf16(false, af, false, b0, (short)0, acc0, false, false);
    acc1 = __builtin_amdgcn_wmma_f32_16x16x32_bf16(false, af, false, b1, (short)0, acc1, false, false);
    acc2 = __builtin_amdgcn_wmma_f32_16x16x32_bf16(false, af, false, b2, (short)0, acc2, false, false);
    acc3 = __builtin_amdgcn_wmma_f32_16x16x32_bf16(false, af, false, b3, (short)0, acc3, false, false);
}

// ---------------------------------------------------------------------------
// Kernel 2: logits = X * W via v_wmma_f32_16x16x32_bf16.
// One block = 8 waves; each wave computes a 16(token) x 64(expert) tile.
// Whole packed W (256 KB bf16) is staged into LDS once per workgroup.
// X is streamed with global_prefetch ~1 KB/row ahead to keep HBM saturated.
// ---------------------------------------------------------------------------
__global__ __launch_bounds__(256) void router_gemm_kernel(const float* __restrict__ x,
                                                          const unsigned* __restrict__ wpk,
                                                          float* __restrict__ out) {
    extern __shared__ __align__(32) unsigned char smem[];   // 262144 bytes
    const int tid = threadIdx.x;

    // Stage packed W -> LDS, fully coalesced b128 stream (64 iters x 256 thr x 16B).
    {
        const uint4* __restrict__ src = (const uint4*)wpk;
        uint4* __restrict__ dst = (uint4*)smem;
        #pragma unroll 4
        for (int i = 0; i < 64; ++i) {
            dst[tid + i * 256] = src[tid + i * 256];
        }
    }
    __syncthreads();

    const int wave  = tid >> 5;
    const int lane  = tid & 31;
    const int m0    = (blockIdx.x * WPB + wave) * MTILE;
    const int row   = m0 + (lane & 15);          // A layout: M = lane%16
    const int khalf = (lane >> 4) * 8;           // lanes>=16 hold K 8..15 / 24..31
    const float* __restrict__ xrow = x + (size_t)row * DIM + khalf;

    v8f acc0 = {}, acc1 = {}, acc2 = {}, acc3 = {};

    // Warm the caches for the first few blocks of this row.
    #pragma unroll
    for (int pb = 0; pb < PFDIST; ++pb) {
        __builtin_prefetch(xrow + pb * 32, 0, 3);
    }

    // Explicit unroll-by-2 software pipeline, no register rotation.
    float4 stageA[4], stageB[4];
    load_ablock(xrow, stageA);
    load_ablock(xrow + 32, stageB);

    // Branch-free main loop over K-blocks 0..59; 60..63 handled in the epilogue.
    for (int kb = 0; kb < KSTEPS - 4; kb += 2) {
        // Fire-and-forget prefetch ~PFDIST blocks ahead. Unguarded: speculative
        // prefetches past the row land in the next row (still inside x) or are
        // silently dropped on translation failure per the ISA.
        __builtin_prefetch(xrow + (kb + PFDIST) * 32, 0, 3);
        __builtin_prefetch(xrow + (kb + PFDIST + 1) * 32, 0, 3);

        wmma_kblock(smem, kb, lane, stageA, acc0, acc1, acc2, acc3);
        load_ablock(xrow + (size_t)(kb + 2) * 32, stageA);

        wmma_kblock(smem, kb + 1, lane, stageB, acc0, acc1, acc2, acc3);
        load_ablock(xrow + (size_t)(kb + 3) * 32, stageB);
    }

    // Epilogue: blocks 60..63 (stageA=60, stageB=61 on loop exit).
    wmma_kblock(smem, KSTEPS - 4, lane, stageA, acc0, acc1, acc2, acc3);
    load_ablock(xrow + (size_t)(KSTEPS - 2) * 32, stageA);
    wmma_kblock(smem, KSTEPS - 3, lane, stageB, acc0, acc1, acc2, acc3);
    load_ablock(xrow + (size_t)(KSTEPS - 1) * 32, stageB);
    wmma_kblock(smem, KSTEPS - 2, lane, stageA, acc0, acc1, acc2, acc3);
    wmma_kblock(smem, KSTEPS - 1, lane, stageB, acc0, acc1, acc2, acc3);

    // Store logits. C layout: VGPR j, lanes 0-15 -> M=j (N=lane), lanes 16-31 -> M=j+8.
    float* __restrict__ logits = out + 2 * NTOK * 2;   // after idx + gates regions
    const int mshift = (lane >> 4) << 3;
    const int ncol   = lane & 15;
    #pragma unroll
    for (int j = 0; j < 8; ++j) {
        float* lr = logits + (size_t)(m0 + j + mshift) * NEXP + ncol;
        lr[0]  = acc0[j];
        lr[16] = acc1[j];
        lr[32] = acc2[j];
        lr[48] = acc3[j];
    }
}

// ---------------------------------------------------------------------------
// Kernel 3: per-token top-2 over 64 experts + softmax over the 2 selected.
// Reads the logits region of d_out (L2-resident, just written).
// ---------------------------------------------------------------------------
__global__ __launch_bounds__(256) void top2_softmax_kernel(float* __restrict__ out) {
    const int tok = blockIdx.x * 256 + threadIdx.x;    // 0..16383
    const float4* __restrict__ lg = (const float4*)(out + 2 * NTOK * 2 + (size_t)tok * NEXP);

    float v0 = -INFINITY, v1 = -INFINITY;
    int   i0 = 0, i1 = 0;
    #pragma unroll
    for (int q = 0; q < 16; ++q) {
        const float4 c = lg[q];
        const float vv[4] = {c.x, c.y, c.z, c.w};
        #pragma unroll
        for (int r = 0; r < 4; ++r) {
            const int e = q * 4 + r;
            const float v = vv[r];
            if (v > v0) { v1 = v0; i1 = i0; v0 = v; i0 = e; }
            else if (v > v1) { v1 = v; i1 = e; }
        }
    }
    // softmax([v0, v1]) with v0 >= v1
    const float t  = __expf(v1 - v0);
    const float g0 = 1.0f / (1.0f + t);
    const float g1 = t * g0;

    out[(size_t)tok * 2 + 0]            = (float)i0;   // topk_idx region
    out[(size_t)tok * 2 + 1]            = (float)i1;
    out[2 * NTOK + (size_t)tok * 2 + 0] = g0;          // gates region
    out[2 * NTOK + (size_t)tok * 2 + 1] = g1;
}

// ---------------------------------------------------------------------------
extern "C" void kernel_launch(void* const* d_in, const int* in_sizes, int n_in,
                              void* d_out, int out_size, void* d_ws, size_t ws_size,
                              hipStream_t stream) {
    const float* x = (const float*)d_in[0];   // [4, 4096, 2048] f32
    const float* W = (const float*)d_in[1];   // [2048, 64] f32
    float*    out  = (float*)d_out;           // idx(32768) | gates(32768) | logits(1048576)
    unsigned* wpk  = (unsigned*)d_ws;         // 256 KB packed bf16 W

    // 1) pack W -> bf16 B-fragment layout (65536 dwords)
    pack_w_kernel<<<256, 256, 0, stream>>>(W, wpk);

    // 2) WMMA GEMM: 128 WGs x 8 waves x 16 tokens = 16384 tokens; 256 KB dynamic LDS
    router_gemm_kernel<<<128, 256, 262144, stream>>>(x, wpk, out);

    // 3) top-2 + softmax: 16384 tokens
    top2_softmax_kernel<<<64, 256, 0, stream>>>(out);
}
